// GGMLLayer_44985487458785
// MI455X (gfx1250) — compile-verified, matrix operand order checked by reading, not run on hardware
//
#include <hip/hip_runtime.h>
#include <cstdint>

// Problem constants (match reference)
#define O_DIM 4096
#define I_DIM 4096
#define R_RNK 16
#define M_TOT 16384          // B*S = 8*2048
#define LORA_SCALE 0.5f
#define Q_ZERO 128.0f

// GEMM tiling
#define BM 128
#define BN 128
#define BK 64

typedef __attribute__((ext_vector_type(16))) __bf16 v16bf;
typedef __attribute__((ext_vector_type(8)))  __bf16 v8bf;
typedef __attribute__((ext_vector_type(8)))  float  v8f;

#if __has_builtin(__builtin_amdgcn_tensor_load_to_lds)
#define HAVE_TDM 1
typedef unsigned int v4u __attribute__((ext_vector_type(4)));
typedef int          v8i __attribute__((ext_vector_type(8)));
typedef int          v4i __attribute__((ext_vector_type(4)));
#endif

__device__ __forceinline__ uint16_t f32_to_bf16_bits(float f) {
    union { float f; uint32_t u; } c; c.f = f;
    uint32_t u = c.u;
    u += 0x7FFFu + ((u >> 16) & 1u);   // round-to-nearest-even
    return (uint16_t)(u >> 16);
}

__device__ __forceinline__ void wait_tensorcnt0() {
#if __has_builtin(__builtin_amdgcn_s_wait_tensorcnt)
    __builtin_amdgcn_s_wait_tensorcnt(0);
#else
    asm volatile("s_wait_tensorcnt 0x0" ::: "memory");
#endif
}

// ---------------------------------------------------------------------------
// Kernel 1: materialize patched weight once:
//   w_bf16[o][i] = scales[o][i/32]*(q[o][i]-128) + 0.5 * sum_r up[o][r]*down[r][i]
// ---------------------------------------------------------------------------
__global__ __launch_bounds__(256)
void prep_weight_kernel(const int* __restrict__ qweight,
                        const float* __restrict__ scales,
                        const float* __restrict__ lora_up,
                        const float* __restrict__ lora_down,
                        uint16_t* __restrict__ w_bf16) {
    const int idx = blockIdx.x * 256 + threadIdx.x;   // 0 .. O*I-1
    const int o = idx >> 12;        // / 4096
    const int i = idx & (I_DIM - 1);

    float patch = 0.0f;
#pragma unroll
    for (int r = 0; r < R_RNK; ++r)
        patch += lora_up[o * R_RNK + r] * lora_down[r * I_DIM + i];

    const float q  = (float)qweight[idx];
    const float sc = scales[o * (I_DIM / 32) + (i >> 5)];
    const float w  = sc * (q - Q_ZERO) + LORA_SCALE * patch;
    w_bf16[idx] = f32_to_bf16_bits(w);
}

// ---------------------------------------------------------------------------
// Kernel 2: out[m][o] = sum_i x[m][i] * w[o][i] + bias[o]
// 256 threads (8 wave32), tile BM x BN, K-step BK=64, double-buffered LDS.
// Weight tiles staged by the Tensor Data Mover (TENSORcnt); x tiles staged
// by vector loads with inline fp32->bf16 conversion.
//
// L2 blocking: blockIdx.x walks N (32 tiles) so consecutive blocks reuse one
// 2 MB x row-tile from L2; the full 32 MB bf16 weight stays L2-resident
// (192 MB), so HBM sees x + w + out roughly once each (~550 MB total).
// ---------------------------------------------------------------------------
__global__ __launch_bounds__(256)
void gemm_bf16_wmma_kernel(const float* __restrict__ x,        // [M_TOT][I_DIM] f32
                           const uint16_t* __restrict__ w,     // [O_DIM][I_DIM] bf16 bits
                           const float* __restrict__ bias,     // [O_DIM]
                           float* __restrict__ out) {          // [M_TOT][O_DIM] f32
    __shared__ __align__(16) uint16_t ldsA[2][BM * BK];   // 16 KB each
    __shared__ __align__(16) uint16_t ldsB[2][BN * BK];   // 16 KB each

    const int tid  = threadIdx.x;
    const int n0   = blockIdx.x * BN;   // N fastest-varying -> x tile reuse in L2
    const int m0   = blockIdx.y * BM;

    const int wid  = tid >> 5;          // wave id 0..7 (wave32)
    const int lane = tid & 31;
    const int half = lane >> 4;         // K-half selector per WMMA layout
    const int l16  = lane & 15;         // M (A) / N (B,C,D) index within frag
    const int wm   = (wid & 3) * 32;    // wave M offset in block tile
    const int wn   = (wid >> 2) * 64;   // wave N offset in block tile

    v8f acc[2][4];
    {
        v8f z = {};
#pragma unroll
        for (int a = 0; a < 2; ++a)
#pragma unroll
            for (int b = 0; b < 4; ++b) acc[a][b] = z;
    }

    // Stage x tile (fp32 -> bf16) into ldsA: BM rows x BK cols
    auto loadA = [&](int buf, int k0) {
#pragma unroll
        for (int j = 0; j < (BM * BK / 4) / 256; ++j) {    // 8 float4 per thread
            const int f   = tid + 256 * j;                 // float4 index
            const int row = f >> (4);                      // BK/4 = 16 float4 per row
            const int c4  = f & 15;
            const float4 v = *(const float4*)(x + (size_t)(m0 + row) * I_DIM + k0 + c4 * 4);
            uint2 p;
            p.x = (uint32_t)f32_to_bf16_bits(v.x) | ((uint32_t)f32_to_bf16_bits(v.y) << 16);
            p.y = (uint32_t)f32_to_bf16_bits(v.z) | ((uint32_t)f32_to_bf16_bits(v.w) << 16);
            *(uint2*)(&ldsA[buf][row * BK + c4 * 4]) = p;
        }
    };

#if HAVE_TDM
    // Tensor Data Mover: DMA a BN x BK bf16 tile of w into ldsB[buf].
    // D# per CDNA5 ISA: group0 = {flags, lds_addr, global_addr,
    // addr_hi|type=2}; group1 = {data_size=2B, tensor dims/strides, tile dims}.
    const uint32_t ldsB_base = (uint32_t)(size_t)(void*)&ldsB[0][0];
    auto tdmLoadB = [&](int buf, int k0) {
        const uint64_t ga = (uint64_t)(size_t)w + (((uint64_t)n0 * I_DIM + (uint64_t)k0) * 2u);
        v4u g0;
        g0[0] = 1u;                                             // count=1, user descriptor
        g0[1] = ldsB_base + (uint32_t)buf * (BN * BK * 2);      // lds_addr (bytes)
        g0[2] = (uint32_t)ga;                                   // global_addr[31:0]
        g0[3] = ((uint32_t)(ga >> 32) & 0x01FFFFFFu)            // global_addr[56:32]
              | 0x80000000u;                                    // type=2 ("image")
        v8i g1;
        g1[0] = (int)(1u << 16);                                // data_size = 1 -> 2 bytes
        g1[1] = (int)(((uint32_t)I_DIM & 0xFFFFu) << 16);       // tensor_dim0[15:0]
        g1[2] = (int)((((uint32_t)I_DIM >> 16) & 0xFFFFu)       // tensor_dim0[31:16]
              | (((uint32_t)O_DIM & 0xFFFFu) << 16));           // tensor_dim1[15:0]
        g1[3] = (int)((((uint32_t)O_DIM >> 16) & 0xFFFFu)       // tensor_dim1[31:16]
              | ((uint32_t)BK << 16));                          // tile_dim0 = BK
        g1[4] = (int)(BN);                                      // tile_dim1 = BN, tile_dim2 = 0
        g1[5] = (int)(I_DIM);                                   // tensor_dim0_stride[31:0]
        g1[6] = 0;
        g1[7] = 0;
        const v4i z4 = {0, 0, 0, 0};                            // groups 2/3 unused (2-D)
        const v8i z8 = {0, 0, 0, 0, 0, 0, 0, 0};
        __builtin_amdgcn_tensor_load_to_lds(g0, g1, z4, z4, z8, 0);
    };
#else
    auto loadBman = [&](int buf, int k0) {
#pragma unroll
        for (int j = 0; j < (BN * BK / 8) / 256; ++j) {
            const int c   = tid + 256 * j;      // 16-byte chunk id
            const int row = c >> 3;             // BK/8 = 8 chunks per row
            const int c8  = c & 7;
            const uint4 v = *(const uint4*)(w + (size_t)(n0 + row) * I_DIM + k0 + c8 * 8);
            *(uint4*)(&ldsB[buf][row * BK + c8 * 8]) = v;
        }
    };
#endif

    // Prologue: stage k-tile 0
#if HAVE_TDM
    if (wid == 0) tdmLoadB(0, 0);
#else
    loadBman(0, 0);
#endif
    loadA(0, 0);
    wait_tensorcnt0();
    __syncthreads();

    const int nk = I_DIM / BK;   // 64 K-steps
    for (int kt = 0; kt < nk; ++kt) {
        const int cur = kt & 1;
        if (kt + 1 < nk) {
#if HAVE_TDM
            if (wid == 0) tdmLoadB(cur ^ 1, (kt + 1) * BK);
#else
            loadBman(cur ^ 1, (kt + 1) * BK);
#endif
            loadA(cur ^ 1, (kt + 1) * BK);
        }

#pragma unroll
        for (int kk = 0; kk < BK / 32; ++kk) {
            // A fragments: lane holds M=l16, K = kk*32 + half*8 (+16)
            v16bf afrag[2];
#pragma unroll
            for (int ms = 0; ms < 2; ++ms) {
                const int row = wm + ms * 16 + l16;
                const v8bf* p = (const v8bf*)(&ldsA[cur][row * BK + kk * 32]);
                const v8bf lo = p[half];
                const v8bf hi = p[half + 2];
                afrag[ms] = __builtin_shufflevector(lo, hi,
                    0, 1, 2, 3, 4, 5, 6, 7, 8, 9, 10, 11, 12, 13, 14, 15);
            }
#pragma unroll
            for (int ns = 0; ns < 4; ++ns) {
                const int row = wn + ns * 16 + l16;   // B lane holds column N=l16
                const v8bf* p = (const v8bf*)(&ldsB[cur][row * BK + kk * 32]);
                const v8bf lo = p[half];
                const v8bf hi = p[half + 2];
                const v16bf bfrag = __builtin_shufflevector(lo, hi,
                    0, 1, 2, 3, 4, 5, 6, 7, 8, 9, 10, 11, 12, 13, 14, 15);
#pragma unroll
                for (int ms = 0; ms < 2; ++ms) {
                    acc[ms][ns] = __builtin_amdgcn_wmma_f32_16x16x32_bf16(
                        /*neg_a=*/false, afrag[ms],
                        /*neg_b=*/false, bfrag,
                        /*c_mod=*/(short)0, acc[ms][ns],
                        /*reuse_a=*/false, /*reuse_b=*/false);
                }
            }
        }

        wait_tensorcnt0();   // DMA for next buffer must land before the barrier
        __syncthreads();
    }

    // Epilogue: C/D layout -> VGPR r holds M = 16*ms + 8*half + r, N = l16
#pragma unroll
    for (int ns = 0; ns < 4; ++ns) {
        const int o  = n0 + wn + ns * 16 + l16;
        const float bv = bias[o];
#pragma unroll
        for (int ms = 0; ms < 2; ++ms) {
            const int mbase = m0 + wm + ms * 16 + half * 8;
#pragma unroll
            for (int r = 0; r < 8; ++r) {
                out[(size_t)(mbase + r) * O_DIM + o] = acc[ms][ns][r] + bv;
            }
        }
    }
}

// ---------------------------------------------------------------------------
// Launch. Workspace requirement: O*I*2 = 32 MB for the bf16 patched weight.
// ---------------------------------------------------------------------------
extern "C" void kernel_launch(void* const* d_in, const int* in_sizes, int n_in,
                              void* d_out, int out_size, void* d_ws, size_t ws_size,
                              hipStream_t stream) {
    const int*   qweight   = (const int*)d_in[0];
    const float* scales    = (const float*)d_in[1];
    const float* lora_up   = (const float*)d_in[2];
    const float* lora_down = (const float*)d_in[3];
    const float* bias      = (const float*)d_in[4];
    const float* x         = (const float*)d_in[5];
    float*       out       = (float*)d_out;

    uint16_t* w_bf16 = (uint16_t*)d_ws;   // needs 4096*4096*2 = 32 MB

    // Pass 1: dequant + LoRA patch -> bf16 weight (once; reused by all tokens)
    const int total = O_DIM * I_DIM;
    prep_weight_kernel<<<total / 256, 256, 0, stream>>>(
        qweight, scales, lora_up, lora_down, w_bf16);

    // Pass 2: WMMA GEMM + bias. grid.x = N tiles (L2 reuse of x row-tiles).
    dim3 grid(O_DIM / BN, M_TOT / BM);   // (32, 128)
    gemm_bf16_wmma_kernel<<<grid, 256, 0, stream>>>(x, w_bf16, bias, out);
}
